// tDBN_2_12841952215183
// MI455X (gfx1250) — matrix operand order repaired and checked by reading, not compiled, not run on hardware
//
#include <hip/hip_runtime.h>
#include <hip/hip_bf16.h>
#include <stdint.h>

// ---------------------------------------------------------------------------
// CDNA5 (gfx1250) implicit-GEMM 3D conv U-Net, bf16 activations/weights,
// f32 accumulate via v_wmma_f32_16x16x32_bf16 (wave32).
//  - activations stored bf16 in ws => A fragments load directly from global
//    in the ISA-documented wave32 A layout (two b128 loads / lane), no LDS.
//  - out-of-bounds / padding taps read from a zeroed guard page instead of
//    branching => straight-line inner loop (no EXEC juggling, no zero-movs).
//  - weights pre-packed per layer into B-fragment bricks (k_packw), so each
//    lane's B fragment is one contiguous 32B load.
//  - one wave = 16 voxels x Cout outputs, NBLK=Cout/16 accumulators => up to
//    8 consecutive WMMAs reuse each A fragment.
// ---------------------------------------------------------------------------

typedef __attribute__((ext_vector_type(16))) __bf16 v16bf;
typedef __attribute__((ext_vector_type(8)))  __bf16 v8bf;
typedef __attribute__((ext_vector_type(8)))  float  v8f;

static __device__ __forceinline__ __bf16 f2bf(float f) {
  union { float f; uint32_t u; } a; a.f = f;
  uint32_t r = (a.u + 0x7FFFu + ((a.u >> 16) & 1u)) >> 16;   // RNE
  union { uint16_t s; __bf16 b; } o; o.s = (uint16_t)r;
  return o.b;
}

struct ConvArgs {
  const __bf16* x;      // activations, bf16, channel-innermost
  const __bf16* wp;     // packed weight bricks
  const __bf16* zpad;   // zeroed guard page (>= Cin bf16 elements)
  const float *g, *b, *m, *v;   // BN params (f32)
  const float* mask;            // occupancy mask (f32)
  void* out;                    // bf16 activations, or f32 for fmap heads
  int Di, Hi, Wi, Do, Ho, Wo;
  int Cin;
  int inCs, outCs, outCo;       // channel strides/offset for concat views
  int KD, KH, KW;
  int stride, padD, padH, padW;
  int transpose;                // conv_transpose(stride2,k=3,VALID) semantics
  int fmap;                     // store as (C*Do, Ho, Wo) f32
};

// Weight pack: brick layout  [tap][kc][nb][lane(32)][j(16)]  with
//   lane = khalf*16 + n ;  element = W[tap][kc*32 + (j<8 ? khalf*8+j : 16+khalf*8+j-8)][nb*16 + n]
__global__ void k_packw(const float* __restrict__ w, __bf16* __restrict__ out,
                        int Cin, int Cout, int ntaps) {
  long i = (long)blockIdx.x * blockDim.x + threadIdx.x;
  long total = (long)ntaps * Cin * Cout;
  if (i >= total) return;
  int j = (int)(i & 15); long t = i >> 4;
  int lane = (int)(t & 31); t >>= 5;
  int nBlk = Cout >> 4, kcN = Cin >> 5;
  int nb = (int)(t % nBlk); t /= nBlk;
  int kc = (int)(t % kcN); int tap = (int)(t / kcN);
  int n = lane & 15, khalf = lane >> 4;
  int k = kc * 32 + (j < 8 ? khalf * 8 + j : 16 + khalf * 8 + (j - 8));
  int c = nb * 16 + n;
  out[i] = f2bf(w[((long)tap * Cin + k) * Cout + c]);
}

template <int NBLK>
__global__ __launch_bounds__(32) void k_conv(ConvArgs A) {
  const int lane  = threadIdx.x;
  const int tileM = blockIdx.x;
  const int HWo   = A.Ho * A.Wo;
  const int Mv    = A.Do * HWo;
  const int n = lane & 15, khalf = lane >> 4;

  // A row per lane: voxel = tileM*16 + (lane&15); lanes 0-15 K 0-7/16-23,
  // lanes 16-31 K 8-15/24-31 (ISA 16-bit A layout).
  const int vox = tileM * 16 + n;
  const int za = vox / HWo;
  const int rv = vox % HWo;
  const int ya = rv / A.Wo, xa = rv % A.Wo;

  v8f acc[NBLK];
#pragma unroll
  for (int b = 0; b < NBLK; ++b) acc[b] = (v8f){0.f,0.f,0.f,0.f,0.f,0.f,0.f,0.f};

  const int kcN = A.Cin >> 5;
  const int nt  = A.KD * A.KH * A.KW;

  for (int tap = 0; tap < nt; ++tap) {
    const int td = tap / (A.KH * A.KW);
    const int rr = tap % (A.KH * A.KW);
    const int th = rr / A.KW, tw = rr % A.KW;

    int zi = 0, yi = 0, xi = 0; bool ok = true;
    if (A.transpose) {            // out[o] += w[t]*x[(o+t-2)/2] (even, in-range)
      int t;
      t = za + td - 2; ok &= (t >= 0) && !(t & 1); zi = t >> 1;
      t = ya + th - 2; ok &= (t >= 0) && !(t & 1); yi = t >> 1;
      t = xa + tw - 2; ok &= (t >= 0) && !(t & 1); xi = t >> 1;
    } else {
      zi = za * A.stride + td - A.padD;
      yi = ya * A.stride + th - A.padH;
      xi = xa * A.stride + tw - A.padW;
      ok = (zi >= 0) && (yi >= 0) && (xi >= 0);
    }
    ok = ok && (zi < A.Di) && (yi < A.Hi) && (xi < A.Wi);
    // zero guard page for invalid taps: loads stay unconditional & in-bounds
    const __bf16* row =
        ok ? (A.x + ((long)(zi * A.Hi + yi) * A.Wi + xi) * A.inCs) : A.zpad;

    const __bf16* wt = A.wp + (long)tap * kcN * (NBLK * 512) + lane * 16;

    for (int kc = 0; kc < kcN; ++kc) {
      const __bf16* ap = row + kc * 32 + khalf * 8;
      v8bf lo = *(const v8bf*)ap;          // K = khalf*8 .. +7
      v8bf hi = *(const v8bf*)(ap + 16);   // K = 16+khalf*8 .. +7
      v16bf av = __builtin_shufflevector(
          lo, hi, 0,1,2,3,4,5,6,7,8,9,10,11,12,13,14,15);
      if (kc + 1 < kcN) __builtin_prefetch(ap + 32, 0, 3);

      const __bf16* wb = wt + (long)kc * (NBLK * 512);
#pragma unroll
      for (int nb = 0; nb < NBLK; ++nb) {
        v16bf bv = *(const v16bf*)(wb + nb * 512);   // contiguous 32B / lane
        acc[nb] = __builtin_amdgcn_wmma_f32_16x16x32_bf16(
                      false, av, false, bv, (short)0, acc[nb], false, false);
      }
    }
  }

  // ---- fused BN + ReLU + mask epilogue ----
  const int mg = khalf * 8;           // C/D rows: M = mg + r, N = lane&15
  float mk[8];
#pragma unroll
  for (int r = 0; r < 8; ++r) {
    int vo = tileM * 16 + mg + r;
    mk[r] = (vo < Mv) ? A.mask[vo] : 0.f;
  }
#pragma unroll
  for (int nb = 0; nb < NBLK; ++nb) {
    const int cl = nb * 16 + n;                       // layer-local channel
    const float scale = A.g[cl] * rsqrtf(A.v[cl] + 1e-3f);
    const float mean = A.m[cl], beta = A.b[cl];
#pragma unroll
    for (int r = 0; r < 8; ++r) {
      const int vo = tileM * 16 + mg + r;
      if (vo >= Mv) continue;
      float yv = (acc[nb][r] - mean) * scale + beta;
      yv = fmaxf(yv, 0.f) * mk[r];
      if (A.fmap) {
        const int z2 = vo / HWo, rem = vo % HWo;
        ((float*)A.out)[((long)cl * A.Do + z2) * HWo + rem] = yv;
      } else {
        ((__bf16*)A.out)[(long)vo * A.outCs + A.outCo + cl] = f2bf(yv);
      }
    }
  }
}

// mask max-pool (VALID), values are {0,1}
__global__ void k_pool(const float* __restrict__ in, float* __restrict__ out,
                       int Di, int Hi, int Wi, int Do, int Ho, int Wo,
                       int kd, int kh, int kw, int s) {
  int i = blockIdx.x * blockDim.x + threadIdx.x;
  int nout = Do * Ho * Wo;
  if (i >= nout) return;
  int z = i / (Ho * Wo); int r = i % (Ho * Wo); int y = r / Wo; int x = r % Wo;
  float mx = 0.f;
  for (int dz = 0; dz < kd; ++dz)
    for (int dy = 0; dy < kh; ++dy)
      for (int dx = 0; dx < kw; ++dx) {
        int zi = z * s + dz, yi = y * s + dy, xi = x * s + dx;
        if (zi < Di && yi < Hi && xi < Wi)
          mx = fmaxf(mx, in[((long)zi * Hi + yi) * Wi + xi]);
      }
  out[i] = mx;
}

// scatter voxel features into dense bf16 NDHWC grid + f32 occupancy mask
__global__ void k_scatter(const float* __restrict__ vf, const int* __restrict__ co,
                          __bf16* __restrict__ dense, float* __restrict__ mask,
                          int NV, int D, int H, int W, int C) {
  int i = blockIdx.x;
  if (i >= NV) return;
  int z = co[i * 4 + 1], y = co[i * 4 + 2], x = co[i * 4 + 3];
  long vox = ((long)z * H + y) * W + x;
  dense[vox * C + threadIdx.x] = f2bf(vf[(long)i * C + threadIdx.x]);
  if (threadIdx.x == 0) mask[vox] = 1.f;
}

// ---------------------------------------------------------------------------
// Host driver.  Param leaves follow JAX pytree flattening (sorted dict keys):
// top level: c0,c1,c2,down1,down2,down3,fm0..3,inp,up10,up21,up32,x0..x3;
// per layer: b,g,m,v,w.  4 data inputs precede the 195 param leaves.
// ---------------------------------------------------------------------------
struct Layer { const float *b, *g, *m, *v, *w; };

extern "C" void kernel_launch(void* const* d_in, const int* in_sizes, int n_in,
                              void* d_out, int out_size, void* d_ws, size_t ws_size,
                              hipStream_t stream) {
  (void)n_in; (void)out_size; (void)ws_size;
  const float* vf   = (const float*)d_in[0];
  const int*  coors = (const int*)d_in[1];
  const int NV = in_sizes[0] / 128;

  const int P0 = 4;
  const int c0_i = P0 + 0,  c1_i = P0 + 20, c2_i = P0 + 40;
  const int d1_i = P0 + 60, d2_i = P0 + 65, d3_i = P0 + 70;
  const int fm0_i = P0 + 75, fm1_i = P0 + 80, fm2_i = P0 + 85, fm3_i = P0 + 90;
  const int inp_i = P0 + 95;
  const int up10_i = P0 + 100, up21_i = P0 + 105, up32_i = P0 + 110;
  const int x0_i = P0 + 115, x1_i = P0 + 135, x2_i = P0 + 155, x3_i = P0 + 175;
  auto L = [&](int idx) -> Layer {
    return Layer{ (const float*)d_in[idx + 0], (const float*)d_in[idx + 1],
                  (const float*)d_in[idx + 2], (const float*)d_in[idx + 3],
                  (const float*)d_in[idx + 4] };
  };

  // dims (input_shape = [160,152,24] -> D=25,H=152,W=160)
  const int D0 = 25, H0 = 152, W0 = 160; const long M0 = (long)D0 * H0 * W0;
  const int D1 = 12, H1 = 75,  W1 = 79;  const long M1 = (long)D1 * H1 * W1;
  const int D2 = 5,  H2 = 37,  W2 = 39;  const long M2 = (long)D2 * H2 * W2;
  const int D3 = 2,  H3 = 18,  W3 = 19;  const long M3 = (long)D3 * H3 * W3;
  const int FD0 = 11, FD1 = 6, FD2 = 3, FD3 = 2;
  const long OFF1 = 128L * FD0 * H0 * W0;
  const long OFF2 = OFF1 + 128L * FD1 * H1 * W1;
  const long OFF3 = OFF2 + 128L * FD2 * H2 * W2;

  // byte-based bump allocator, 64B aligned
  char* base = (char*)d_ws; size_t off = 0;
  auto alloc = [&](size_t bytes) {
    off = (off + 63) & ~(size_t)63;
    char* p = base + off; off += bytes; return p;
  };

  __bf16* dense = (__bf16*)alloc(M0 * 128 * 2);   // recycled after 'inp' conv
  float*  mask0 = (float*)alloc(M0 * 4);
  __bf16* p0a   = (__bf16*)alloc(M0 * 32 * 2);
  __bf16* p0b   = dense;                 // alias: dense dead after inp conv
  __bf16* xf0   = dense + M0 * 32;       // 64ch concat view inside dense region
  float*  mask1 = (float*)alloc(M1 * 4);
  __bf16* p1a = (__bf16*)alloc(M1 * 64 * 2); __bf16* p1b = (__bf16*)alloc(M1 * 64 * 2);
  __bf16* xf1 = (__bf16*)alloc(M1 * 128 * 2);
  float*  mask2 = (float*)alloc(M2 * 4);
  __bf16* p2a = (__bf16*)alloc(M2 * 96 * 2); __bf16* p2b = (__bf16*)alloc(M2 * 96 * 2);
  __bf16* xf2 = (__bf16*)alloc(M2 * 192 * 2);
  float*  mask3 = (float*)alloc(M3 * 4);
  __bf16* x3b = (__bf16*)alloc(M3 * 128 * 2); __bf16* p3b = (__bf16*)alloc(M3 * 128 * 2);
  float* fmk0 = (float*)alloc((long)FD0 * H0 * W0 * 4);
  float* fmk1 = (float*)alloc((long)FD1 * H1 * W1 * 4);
  float* fmk2 = (float*)alloc((long)FD2 * H2 * W2 * 4);
  __bf16* wpack = (__bf16*)alloc(27L * 192 * 128 * 2);   // max brick set
  __bf16* zpad  = (__bf16*)alloc(512);                   // zero guard page
  float* dout = (float*)d_out;

  auto conv = [&](const __bf16* x, int ics, Layer P, const float* mask,
                  void* out, int ocs, int oco,
                  int Di, int Hi, int Wi, int Do, int Ho, int Wo,
                  int Cin, int Cout, int KD, int KH, int KW,
                  int st, int pd, int ph, int pw2, int tr, int fm) {
    const int ntaps = KD * KH * KW;
    const long total = (long)ntaps * Cin * Cout;
    k_packw<<<dim3((unsigned)((total + 255) / 256)), 256, 0, stream>>>(
        P.w, wpack, Cin, Cout, ntaps);
    ConvArgs A;
    A.x = x; A.wp = wpack; A.zpad = zpad;
    A.g = P.g; A.b = P.b; A.m = P.m; A.v = P.v;
    A.mask = mask; A.out = out;
    A.Di = Di; A.Hi = Hi; A.Wi = Wi; A.Do = Do; A.Ho = Ho; A.Wo = Wo;
    A.Cin = Cin; A.inCs = ics; A.outCs = ocs; A.outCo = oco;
    A.KD = KD; A.KH = KH; A.KW = KW; A.stride = st;
    A.padD = pd; A.padH = ph; A.padW = pw2; A.transpose = tr; A.fmap = fm;
    const int Mv = Do * Ho * Wo;
    dim3 grid((unsigned)((Mv + 15) / 16));
    switch (Cout >> 4) {
      case 2: k_conv<2><<<grid, 32, 0, stream>>>(A); break;
      case 4: k_conv<4><<<grid, 32, 0, stream>>>(A); break;
      case 6: k_conv<6><<<grid, 32, 0, stream>>>(A); break;
      default: k_conv<8><<<grid, 32, 0, stream>>>(A); break;
    }
  };
  auto pool = [&](const float* in, float* out, int Di, int Hi, int Wi,
                  int Do, int Ho, int Wo, int kd, int kh, int kw, int s) {
    int nout = Do * Ho * Wo;
    k_pool<<<(nout + 255) / 256, 256, 0, stream>>>(in, out, Di, Hi, Wi,
                                                   Do, Ho, Wo, kd, kh, kw, s);
  };

  // ---- init + scatter ----
  hipMemsetAsync(zpad, 0, 512, stream);
  hipMemsetAsync(dense, 0, (size_t)M0 * 128 * 2, stream);
  hipMemsetAsync(mask0, 0, (size_t)M0 * 4, stream);
  k_scatter<<<NV, 128, 0, stream>>>(vf, coors, dense, mask0, NV, D0, H0, W0, 128);

  // ---- stem + x0 chain (ends inside xf0[:, :32]) ----
  conv(dense, 128, L(inp_i), mask0, p0a, 32, 0, D0,H0,W0, D0,H0,W0, 128,32, 3,3,3, 1,1,1,1, 0,0);
  conv(p0a, 32, L(x0_i+0),  mask0, p0b, 32, 0, D0,H0,W0, D0,H0,W0, 32,32, 3,3,3, 1,1,1,1, 0,0);
  conv(p0b, 32, L(x0_i+5),  mask0, p0a, 32, 0, D0,H0,W0, D0,H0,W0, 32,32, 3,3,3, 1,1,1,1, 0,0);
  conv(p0a, 32, L(x0_i+10), mask0, p0b, 32, 0, D0,H0,W0, D0,H0,W0, 32,32, 3,3,3, 1,1,1,1, 0,0);
  conv(p0b, 32, L(x0_i+15), mask0, xf0, 64, 0, D0,H0,W0, D0,H0,W0, 32,32, 3,3,3, 1,1,1,1, 0,0);

  // ---- down1 + x1 chain (ends inside xf1[:, :64]) ----
  pool(mask0, mask1, D0,H0,W0, D1,H1,W1, 3,3,3, 2);
  conv(xf0, 64, L(d1_i), mask1, p1a, 64, 0, D0,H0,W0, D1,H1,W1, 32,64, 3,3,3, 2,0,0,0, 0,0);
  conv(p1a, 64, L(x1_i+0),  mask1, p1b, 64, 0, D1,H1,W1, D1,H1,W1, 64,64, 3,3,3, 1,1,1,1, 0,0);
  conv(p1b, 64, L(x1_i+5),  mask1, p1a, 64, 0, D1,H1,W1, D1,H1,W1, 64,64, 3,3,3, 1,1,1,1, 0,0);
  conv(p1a, 64, L(x1_i+10), mask1, p1b, 64, 0, D1,H1,W1, D1,H1,W1, 64,64, 3,3,3, 1,1,1,1, 0,0);
  conv(p1b, 64, L(x1_i+15), mask1, xf1, 128, 0, D1,H1,W1, D1,H1,W1, 64,64, 3,3,3, 1,1,1,1, 0,0);

  // ---- down2 + x2 chain (ends inside xf2[:, :96]) ----
  pool(mask1, mask2, D1,H1,W1, D2,H2,W2, 3,3,3, 2);
  conv(xf1, 128, L(d2_i), mask2, p2a, 96, 0, D1,H1,W1, D2,H2,W2, 64,96, 3,3,3, 2,0,0,0, 0,0);
  conv(p2a, 96, L(x2_i+0),  mask2, p2b, 96, 0, D2,H2,W2, D2,H2,W2, 96,96, 3,3,3, 1,1,1,1, 0,0);
  conv(p2b, 96, L(x2_i+5),  mask2, p2a, 96, 0, D2,H2,W2, D2,H2,W2, 96,96, 3,3,3, 1,1,1,1, 0,0);
  conv(p2a, 96, L(x2_i+10), mask2, p2b, 96, 0, D2,H2,W2, D2,H2,W2, 96,96, 3,3,3, 1,1,1,1, 0,0);
  conv(p2b, 96, L(x2_i+15), mask2, xf2, 192, 0, D2,H2,W2, D2,H2,W2, 96,96, 3,3,3, 1,1,1,1, 0,0);

  // ---- down3 + x3 chain ----
  pool(mask2, mask3, D2,H2,W2, D3,H3,W3, 3,3,3, 2);
  conv(xf2, 192, L(d3_i), mask3, x3b, 128, 0, D2,H2,W2, D3,H3,W3, 96,128, 3,3,3, 2,0,0,0, 0,0);
  conv(x3b, 128, L(x3_i+0),  mask3, p3b, 128, 0, D3,H3,W3, D3,H3,W3, 128,128, 3,3,3, 1,1,1,1, 0,0);
  conv(p3b, 128, L(x3_i+5),  mask3, x3b, 128, 0, D3,H3,W3, D3,H3,W3, 128,128, 3,3,3, 1,1,1,1, 0,0);
  conv(x3b, 128, L(x3_i+10), mask3, p3b, 128, 0, D3,H3,W3, D3,H3,W3, 128,128, 3,3,3, 1,1,1,1, 0,0);
  conv(p3b, 128, L(x3_i+15), mask3, x3b, 128, 0, D3,H3,W3, D3,H3,W3, 128,128, 3,3,3, 1,1,1,1, 0,0);

  // ---- up32 + c2 chain (final in p2b) ----
  conv(x3b, 128, L(up32_i), mask2, xf2, 192, 96, D3,H3,W3, D2,H2,W2, 128,96, 3,3,3, 1,0,0,0, 1,0);
  conv(xf2, 192, L(c2_i+0),  mask2, p2a, 96, 0, D2,H2,W2, D2,H2,W2, 192,96, 3,3,3, 1,1,1,1, 0,0);
  conv(p2a, 96,  L(c2_i+5),  mask2, p2b, 96, 0, D2,H2,W2, D2,H2,W2, 96,96, 3,3,3, 1,1,1,1, 0,0);
  conv(p2b, 96,  L(c2_i+10), mask2, p2a, 96, 0, D2,H2,W2, D2,H2,W2, 96,96, 3,3,3, 1,1,1,1, 0,0);
  conv(p2a, 96,  L(c2_i+15), mask2, p2b, 96, 0, D2,H2,W2, D2,H2,W2, 96,96, 3,3,3, 1,1,1,1, 0,0);

  // ---- up21 + c1 chain (final in p1b) ----
  conv(p2b, 96, L(up21_i), mask1, xf1, 128, 64, D2,H2,W2, D1,H1,W1, 96,64, 3,3,3, 1,0,0,0, 1,0);
  conv(xf1, 128, L(c1_i+0),  mask1, p1a, 64, 0, D1,H1,W1, D1,H1,W1, 128,64, 3,3,3, 1,1,1,1, 0,0);
  conv(p1a, 64,  L(c1_i+5),  mask1, p1b, 64, 0, D1,H1,W1, D1,H1,W1, 64,64, 3,3,3, 1,1,1,1, 0,0);
  conv(p1b, 64,  L(c1_i+10), mask1, p1a, 64, 0, D1,H1,W1, D1,H1,W1, 64,64, 3,3,3, 1,1,1,1, 0,0);
  conv(p1a, 64,  L(c1_i+15), mask1, p1b, 64, 0, D1,H1,W1, D1,H1,W1, 64,64, 3,3,3, 1,1,1,1, 0,0);

  // ---- up10 + c0 chain (final in p0b) ----
  conv(p1b, 64, L(up10_i), mask0, xf0, 64, 32, D1,H1,W1, D0,H0,W0, 64,32, 3,3,3, 1,0,0,0, 1,0);
  conv(xf0, 64, L(c0_i+0),  mask0, p0a, 32, 0, D0,H0,W0, D0,H0,W0, 64,32, 3,3,3, 1,1,1,1, 0,0);
  conv(p0a, 32, L(c0_i+5),  mask0, p0b, 32, 0, D0,H0,W0, D0,H0,W0, 32,32, 3,3,3, 1,1,1,1, 0,0);
  conv(p0b, 32, L(c0_i+10), mask0, p0a, 32, 0, D0,H0,W0, D0,H0,W0, 32,32, 3,3,3, 1,1,1,1, 0,0);
  conv(p0a, 32, L(c0_i+15), mask0, p0b, 32, 0, D0,H0,W0, D0,H0,W0, 32,32, 3,3,3, 1,1,1,1, 0,0);

  // ---- fmap heads: (kd,1,1) VALID conv + BN/ReLU/pooled-mask, f32 transpose store ----
  pool(mask0, fmk0, D0,H0,W0, FD0,H0,W0, 15,1,1, 1);
  conv(p0b, 32, L(fm0_i), fmk0, dout + 0, 0, 0, D0,H0,W0, FD0,H0,W0, 32,128, 15,1,1, 1,0,0,0, 0,1);
  pool(mask1, fmk1, D1,H1,W1, FD1,H1,W1, 7,1,1, 1);
  conv(p1b, 64, L(fm1_i), fmk1, dout + OFF1, 0, 0, D1,H1,W1, FD1,H1,W1, 64,128, 7,1,1, 1,0,0,0, 0,1);
  pool(mask2, fmk2, D2,H2,W2, FD2,H2,W2, 3,1,1, 1);
  conv(p2b, 96, L(fm2_i), fmk2, dout + OFF2, 0, 0, D2,H2,W2, FD2,H2,W2, 96,128, 3,1,1, 1,0,0,0, 0,1);
  conv(x3b, 128, L(fm3_i), mask3, dout + OFF3, 0, 0, D3,H3,W3, FD3,H3,W3, 128,128, 1,1,1, 1,0,0,0, 0,1);
}
